// CausalSelectiveSelfAttention_81355270521192
// MI455X (gfx1250) — compile-verified, matrix operand order checked by reading, not run on hardware
//
#include <hip/hip_runtime.h>
#include <hip/hip_bf16.h>

// Problem constants (from reference): B=2, T=2048, C=768, H=12, HD=64
#define Bb  2
#define Tt  2048
#define Cc  768
#define Hh  12
#define HDd 64

typedef __bf16 bf16;
typedef bf16  bf16x8  __attribute__((ext_vector_type(8)));
typedef bf16  bf16x16 __attribute__((ext_vector_type(16)));
typedef float v8f     __attribute__((ext_vector_type(8)));

__device__ inline v8f vzero() {
  v8f z;
#pragma unroll
  for (int i = 0; i < 8; ++i) z[i] = 0.0f;
  return z;
}

// Load one 16x32 bf16 WMMA fragment row-slice for this lane.
// A layout (ISA 7.12.2): lane&15 = matrix row, lanes 0-15 hold K {0..7, 16..23},
// lanes 16-31 hold K {8..15, 24..31}. Two contiguous 16B loads.
__device__ inline bf16x16 load_frag(const bf16* p, int half) {
  bf16x8 lo = *(const bf16x8*)(p + half * 8);
  bf16x8 hi = *(const bf16x8*)(p + 16 + half * 8);
  bf16x16 r;
#pragma unroll
  for (int i = 0; i < 8; ++i) { r[i] = lo[i]; r[i + 8] = hi[i]; }
  return r;
}

// CDNA5 async global -> LDS copy, 16 bytes per lane, tracked by ASYNCcnt.
__device__ inline void async_b128(unsigned lds_addr, const bf16* g) {
  asm volatile("global_load_async_to_lds_b128 %0, %1, off"
               :: "v"(lds_addr), "v"(g) : "memory");
}

// ---------------------------------------------------------------------------
// GEMM: Out[M,N] = A[M,K] * W[N,K]^T + bias. bf16 operands, f32 accumulate.
// Block tile 64(M) x 128(N), BK=64 (2 WMMA k-steps). 8 waves = 2x4, each 32x32.
// Double-buffered LDS filled with global_load_async_to_lds_b128; loads for
// stage k+1 overlap WMMA of stage k; drained with s_wait_asynccnt.
// M%64==0, N%128==0, K%64==0 for all uses here.
// ---------------------------------------------------------------------------
__global__ __launch_bounds__(256) void gemm_bias_wmma(
    const bf16* __restrict__ A, long batchA, int ldA,
    const bf16* __restrict__ W, long batchB, int ldB,
    const float* __restrict__ bias,
    float* __restrict__ Out, long batchO, int ldO, int K) {
  constexpr int BM = 64, BN = 128, BK = 64, LDK = BK + 8;  // 72 (pad vs bank conflicts)
  __shared__ __align__(16) bf16 As[2][BM][LDK];
  __shared__ __align__(16) bf16 Bs[2][BN][LDK];

  A   += (long)blockIdx.z * batchA;
  W   += (long)blockIdx.z * batchB;
  Out += (long)blockIdx.z * batchO;

  const int tid  = threadIdx.x;
  const int wid  = tid >> 5;
  const int lane = tid & 31;
  const int half = lane >> 4;
  const int lrow = lane & 15;
  const int wm   = wid >> 2;      // 0..1
  const int wn   = wid & 3;       // 0..3
  const int gm   = blockIdx.x * BM;
  const int gn   = blockIdx.y * BN;

  // Async staging: 16B (8 bf16) chunks. A: 64 rows x 8 chunks = 512 -> 2/thread.
  // B: 128 rows x 8 chunks = 1024 -> 4/thread.  6 async instr per wave per stage.
  auto issue = [&](int buf, int kk) {
#pragma unroll
    for (int e = 0; e < 2; ++e) {
      const int c = tid + e * 256;
      const int r = c >> 3, col = (c & 7) * 8;
      async_b128((unsigned)(uintptr_t)&As[buf][r][col],
                 A + (long)(gm + r) * ldA + kk + col);
    }
#pragma unroll
    for (int e = 0; e < 4; ++e) {
      const int c = tid + e * 256;
      const int r = c >> 3, col = (c & 7) * 8;
      async_b128((unsigned)(uintptr_t)&Bs[buf][r][col],
                 W + (long)(gn + r) * ldB + kk + col);
    }
  };

  v8f acc[2][2];
#pragma unroll
  for (int i = 0; i < 2; ++i)
#pragma unroll
    for (int j = 0; j < 2; ++j) acc[i][j] = vzero();

  issue(0, 0);
  int cur = 0;
  for (int kk = 0; kk < K; kk += BK) {
    const bool more = (kk + BK) < K;
    if (more) {
      issue(cur ^ 1, kk + BK);                      // overlap with this stage's WMMA
      asm volatile("s_wait_asynccnt 6" ::: "memory");  // retire current buffer only
    } else {
      asm volatile("s_wait_asynccnt 0" ::: "memory");
    }
    __syncthreads();                                // cur buffer visible to all waves

#pragma unroll
    for (int ks = 0; ks < 2; ++ks) {
      bf16x16 a0 = load_frag(&As[cur][wm * 32 + lrow][ks * 32], half);
      bf16x16 a1 = load_frag(&As[cur][wm * 32 + 16 + lrow][ks * 32], half);
      bf16x16 b0 = load_frag(&Bs[cur][wn * 32 + lrow][ks * 32], half);
      bf16x16 b1 = load_frag(&Bs[cur][wn * 32 + 16 + lrow][ks * 32], half);
      acc[0][0] = __builtin_amdgcn_wmma_f32_16x16x32_bf16(false, a0, false, b0, (short)0, acc[0][0], false, false);
      acc[0][1] = __builtin_amdgcn_wmma_f32_16x16x32_bf16(false, a0, false, b1, (short)0, acc[0][1], false, false);
      acc[1][0] = __builtin_amdgcn_wmma_f32_16x16x32_bf16(false, a1, false, b0, (short)0, acc[1][0], false, false);
      acc[1][1] = __builtin_amdgcn_wmma_f32_16x16x32_bf16(false, a1, false, b1, (short)0, acc[1][1], false, false);
    }
    __syncthreads();                                // all waves done reading cur
    cur ^= 1;
  }

#pragma unroll
  for (int ms = 0; ms < 2; ++ms)
#pragma unroll
    for (int ns = 0; ns < 2; ++ns) {
      const int col = gn + wn * 32 + ns * 16 + lrow;
      const float bv = bias ? bias[col] : 0.0f;
#pragma unroll
      for (int v = 0; v < 8; ++v) {
        const int row = gm + wm * 32 + ms * 16 + v + half * 8;
        Out[(long)row * ldO + col] = acc[ms][ns][v] + bv;
      }
    }
}

// ---------------------------------------------------------------------------
__global__ void f32_to_bf16_kernel(const float* __restrict__ in,
                                   bf16* __restrict__ outp, long n) {
  const long i = (long)blockIdx.x * blockDim.x + threadIdx.x;
  if (i < n) outp[i] = (bf16)in[i];
}

// ---------------------------------------------------------------------------
// Split qkv[B,T,3C] into:
//   qb  bf16 [B,H,T,64]   (attention Q, row-contiguous in d)
//   kc  bf16 [B,T,768]    (K concat: attention B-frags AND S-GEMM B matrix)
//   vT  bf16 [B,H,64,T]   (V transposed: contiguous B-frag loads for P@V)
//   qs  bf16 [B,T,768]    (Q scaled by sel_w[h]*scale: S-GEMM A matrix)
// ---------------------------------------------------------------------------
__global__ void pack_qkv(const float* __restrict__ qkv,
                         const float* __restrict__ sel_w,
                         bf16* __restrict__ qb, bf16* __restrict__ kc,
                         bf16* __restrict__ vT, bf16* __restrict__ qs) {
  const int idx = blockIdx.x * blockDim.x + threadIdx.x;
  if (idx >= Bb * Tt * Cc) return;
  const int c = idx % Cc;
  const int t = (idx / Cc) % Tt;
  const int b = idx / (Cc * Tt);
  const int h = c >> 6;
  const int d = c & 63;
  const float* base = qkv + (long)(b * Tt + t) * (3 * Cc);
  const float q = base[c];
  const float k = base[Cc + c];
  const float v = base[2 * Cc + c];
  qb[(((long)(b * Hh + h) * Tt) + t) * HDd + d] = (bf16)q;
  kc[((long)b * Tt + t) * Cc + c] = (bf16)k;
  vT[(((long)(b * Hh + h) * HDd) + d) * Tt + t] = (bf16)v;
  qs[((long)b * Tt + t) * Cc + c] = (bf16)(q * sel_w[h] * 0.125f);  // 1/sqrt(64)
}

// ---------------------------------------------------------------------------
// In-place: S[b,i,j] (raw sel logits) -> FF_shifted[b,i,j] = sum_{i'<i} S'(i',j)
// where S'(i',j) = relu(S) if (j < i' && j != 0) else 0  (causal + diag + BOS).
// One thread per column j: coalesced across threads at every row step.
// ---------------------------------------------------------------------------
__global__ void ff_cumsum(float* __restrict__ S) {
  const int j = blockIdx.x * blockDim.x + threadIdx.x;
  const int b = blockIdx.y;
  float* col = S + (long)b * Tt * Tt + j;
  float acc = 0.0f;
  for (int i = 0; i < Tt; ++i) {
    const float sraw = col[(long)i * Tt];
    const float s = (j < i && j > 0) ? fmaxf(sraw, 0.0f) : 0.0f;
    col[(long)i * Tt] = acc;   // FF_shifted[i] uses rows < i
    acc += s;
  }
}

// ---------------------------------------------------------------------------
// Flash-style causal attention with FF_shifted bias.
// Block = 4 waves; wave w handles rows [i0, i0+16) of one (b,h).
// Streams 32-column K/V tiles (causal early exit), online softmax with
// half-wave shuffles, P transposed C->A layout through per-wave LDS.
// Output y is bf16 so the projection GEMM consumes it directly.
// ---------------------------------------------------------------------------
__global__ __launch_bounds__(128) void attn_kernel(
    const bf16* __restrict__ qb, const bf16* __restrict__ kc,
    const bf16* __restrict__ vT, const float* __restrict__ FF,
    bf16* __restrict__ y) {
  __shared__ __align__(16) bf16 Pl[4][16][32];
  const int tid  = threadIdx.x;
  const int wid  = tid >> 5;
  const int lane = tid & 31;
  const int half = lane >> 4;
  const int lrow = lane & 15;
  const int bh = blockIdx.y;
  const int b  = bh / Hh;
  const int h  = bh - b * Hh;
  const int i0 = blockIdx.x * 64 + wid * 16;

  const bf16* Q   = qb + (long)bh * Tt * HDd;
  const bf16* Kc  = kc + (long)b * Tt * Cc + h * HDd;
  const bf16* Vt  = vT + (long)bh * HDd * Tt;
  const float* FFb = FF + (long)b * Tt * Tt;
  const float NEG = -__builtin_inff();

  bf16x16 aQ[2];
#pragma unroll
  for (int ks = 0; ks < 2; ++ks)
    aQ[ks] = load_frag(Q + (long)(i0 + lrow) * HDd + ks * 32, half);

  float mrow[8], lsum[8];
  v8f o[4];
#pragma unroll
  for (int v = 0; v < 8; ++v) { mrow[v] = NEG; lsum[v] = 0.0f; }
#pragma unroll
  for (int ns = 0; ns < 4; ++ns) o[ns] = vzero();

  for (int j0 = 0; j0 < i0 + 16; j0 += 32) {
    // S tile (16 rows x 32 cols) = Q(16x64) @ K^T
    v8f c[2];
    c[0] = vzero(); c[1] = vzero();
#pragma unroll
    for (int ns = 0; ns < 2; ++ns) {
      bf16x16 bk0 = load_frag(Kc + (long)(j0 + ns * 16 + lrow) * Cc + 0, half);
      bf16x16 bk1 = load_frag(Kc + (long)(j0 + ns * 16 + lrow) * Cc + 32, half);
      c[ns] = __builtin_amdgcn_wmma_f32_16x16x32_bf16(false, aQ[0], false, bk0, (short)0, c[ns], false, false);
      c[ns] = __builtin_amdgcn_wmma_f32_16x16x32_bf16(false, aQ[1], false, bk1, (short)0, c[ns], false, false);
    }
    // scale, FF bias, causal mask, online softmax
#pragma unroll
    for (int v = 0; v < 8; ++v) {
      const int row = i0 + v + half * 8;
      float s0 = c[0][v] * 0.125f - FFb[(long)row * Tt + (j0 + lrow)];
      float s1 = c[1][v] * 0.125f - FFb[(long)row * Tt + (j0 + 16 + lrow)];
      if (j0 + lrow > row)      s0 = NEG;
      if (j0 + 16 + lrow > row) s1 = NEG;
      float mx = fmaxf(s0, s1);
#pragma unroll
      for (int off = 1; off < 16; off <<= 1) mx = fmaxf(mx, __shfl_xor(mx, off, 32));
      const float nm   = fmaxf(mrow[v], mx);
      const float corr = __expf(mrow[v] - nm);
      const float p0 = __expf(s0 - nm);
      const float p1 = __expf(s1 - nm);
      float ps = p0 + p1;
#pragma unroll
      for (int off = 1; off < 16; off <<= 1) ps += __shfl_xor(ps, off, 32);
      lsum[v] = lsum[v] * corr + ps;
      mrow[v] = nm;
#pragma unroll
      for (int ns = 0; ns < 4; ++ns) o[ns][v] *= corr;
      const int m = v + half * 8;
      Pl[wid][m][lrow]      = (bf16)p0;
      Pl[wid][m][16 + lrow] = (bf16)p1;
    }
    // same-wave LDS is in-order; fence the compiler then reload P in A layout
    asm volatile("s_wait_dscnt 0" ::: "memory");
    bf16x16 aP = load_frag(&Pl[wid][lrow][0], half);
#pragma unroll
    for (int ns = 0; ns < 4; ++ns) {
      bf16x16 bV = load_frag(Vt + (long)(ns * 16 + lrow) * Tt + j0, half);
      o[ns] = __builtin_amdgcn_wmma_f32_16x16x32_bf16(false, aP, false, bV, (short)0, o[ns], false, false);
    }
  }

#pragma unroll
  for (int ns = 0; ns < 4; ++ns)
#pragma unroll
    for (int v = 0; v < 8; ++v) {
      const int row = i0 + v + half * 8;
      y[((long)b * Tt + row) * Cc + h * HDd + ns * 16 + lrow] =
          (bf16)(o[ns][v] / lsum[v]);
    }
}

// ---------------------------------------------------------------------------
extern "C" void kernel_launch(void* const* d_in, const int* in_sizes, int n_in,
                              void* d_out, int out_size, void* d_ws, size_t ws_size,
                              hipStream_t stream) {
  (void)in_sizes; (void)n_in; (void)out_size; (void)ws_size;
  const float* x      = (const float*)d_in[0];
  const float* w_attn = (const float*)d_in[1];
  const float* b_attn = (const float*)d_in[2];
  const float* w_proj = (const float*)d_in[3];
  const float* b_proj = (const float*)d_in[4];
  const float* sel_w  = (const float*)d_in[5];
  float* out = (float*)d_out;

  char* p = (char*)d_ws;
  auto alloc = [&](size_t bytes) {
    char* r = p;
    p += (bytes + 255) & ~(size_t)255;
    return r;
  };
  const long nX  = (long)Bb * Tt * Cc;       // 3,145,728
  const long nWA = (long)3 * Cc * Cc;        // 1,769,472
  const long nWP = (long)Cc * Cc;            //   589,824

  float* qkv = (float*)alloc((size_t)Bb * Tt * 3 * Cc * 4);  // 37.7 MB
  float* Sff = (float*)alloc((size_t)Bb * Tt * Tt * 4);      // 33.6 MB
  bf16*  xb  = (bf16*)alloc((size_t)nX * 2);
  bf16*  wab = (bf16*)alloc((size_t)nWA * 2);
  bf16*  wpb = (bf16*)alloc((size_t)nWP * 2);
  bf16*  qb  = (bf16*)alloc((size_t)nX * 2);
  bf16*  kc  = (bf16*)alloc((size_t)nX * 2);
  bf16*  vT  = (bf16*)alloc((size_t)nX * 2);
  bf16*  qsb = (bf16*)alloc((size_t)nX * 2);
  bf16*  ybb = (bf16*)alloc((size_t)nX * 2);

  // 0) one-shot f32 -> bf16 conversions (activations + weights)
  f32_to_bf16_kernel<<<(nX + 255) / 256, 256, 0, stream>>>(x, xb, nX);
  f32_to_bf16_kernel<<<(nWA + 255) / 256, 256, 0, stream>>>(w_attn, wab, nWA);
  f32_to_bf16_kernel<<<(nWP + 255) / 256, 256, 0, stream>>>(w_proj, wpb, nWP);

  // 1) qkv = x @ w_attn^T + b_attn            [4096 x 2304, K=768]
  gemm_bias_wmma<<<dim3((Bb * Tt) / 64, (3 * Cc) / 128, 1), 256, 0, stream>>>(
      xb, 0, Cc, wab, 0, Cc, b_attn, qkv, 0, 3 * Cc, Cc);

  // 2) repack q/k/v into attention-friendly layouts
  pack_qkv<<<(Bb * Tt * Cc + 255) / 256, 256, 0, stream>>>(qkv, sel_w, qb, kc, vT, qsb);

  // 3) S raw logits = qs @ kc^T per batch     [2048 x 2048, K=768] x B
  gemm_bias_wmma<<<dim3(Tt / 64, Tt / 128, Bb), 256, 0, stream>>>(
      qsb, (long)Tt * Cc, Cc, kc, (long)Tt * Cc, Cc, nullptr,
      Sff, (long)Tt * Tt, Tt, Cc);

  // 4) mask/relu + shifted column cumsum -> FF_shifted (in place)
  ff_cumsum<<<dim3(Tt / 256, Bb), 256, 0, stream>>>(Sff);

  // 5) flash attention with FF bias -> ybb bf16 [B,T,768]
  attn_kernel<<<dim3(Tt / 64, Bb * Hh), 128, 0, stream>>>(qb, kc, vT, Sff, ybb);

  // 6) out = y @ w_proj^T + b_proj            [4096 x 768, K=768]
  gemm_bias_wmma<<<dim3((Bb * Tt) / 64, Cc / 128, 1), 256, 0, stream>>>(
      ybb, 0, Cc, wpb, 0, Cc, b_proj, out, 0, Cc, Cc);
}
